// CapsNet_12214886990603
// MI455X (gfx1250) — matrix operand order repaired and checked by reading, not compiled, not run on hardware
//
#include <hip/hip_runtime.h>
#include <hip/hip_bf16.h>
#include <math.h>

typedef __attribute__((ext_vector_type(16))) __bf16 v16bf;
typedef __attribute__((ext_vector_type(8)))  __bf16 v8bf;
typedef __attribute__((ext_vector_type(8)))  float  v8f;
typedef __attribute__((ext_vector_type(4)))  unsigned int u32x4;
typedef __attribute__((ext_vector_type(8)))  int        i32x8;
typedef __attribute__((ext_vector_type(4)))  int        i32x4;

#if defined(__has_builtin)
#if __has_builtin(__builtin_amdgcn_tensor_load_to_lds) && __has_builtin(__builtin_amdgcn_s_wait_tensorcnt)
#define HAVE_TDM 1
#endif
#endif
#ifndef HAVE_TDM
#define HAVE_TDM 0
#endif

// ---------------------------------------------------------------- helpers
__device__ __forceinline__ unsigned short f2bf(float f) {
  unsigned int u = __float_as_uint(f);
  unsigned int r = (u + 0x7FFFu + ((u >> 16) & 1u)) >> 16;  // RNE
  return (unsigned short)r;
}

// ---------------------------------------------------------------- conv1
// data (256,1,28,28) f32 ; w1 (256,1,9,9) ; b1 (256)
// out  x1 (256,256,20,20) bf16, ReLU applied
__global__ void conv1_kernel(const float* __restrict__ x,
                             const float* __restrict__ w1,
                             const float* __restrict__ b1,
                             unsigned short* __restrict__ x1) {
  int idx = blockIdx.x * 256 + threadIdx.x;            // 26,214,400 total
  if (idx >= 256 * 256 * 400) return;
  int ox = idx % 20;
  int t  = idx / 20;
  int oy = t % 20;  t /= 20;
  int oc = t % 256;
  int b  = t / 256;
  const float* xin = x + b * 784;
  const float* w   = w1 + oc * 81;
  float acc = b1[oc];
#pragma unroll
  for (int ky = 0; ky < 9; ++ky) {
#pragma unroll
    for (int kx = 0; kx < 9; ++kx) {
      acc = fmaf(xin[(oy + ky) * 28 + ox + kx], w[ky * 9 + kx], acc);
    }
  }
  x1[idx] = f2bf(fmaxf(acc, 0.0f));
}

// ---------------------------------------------------------------- pack w2
// w2 (256oc,256ic,9,9) f32 -> bf16 B-fragments: [kt 648][nt 16][lane 32][j 16]
// B (32x16, 16-bit) lane layout: lane<16 -> N=lane,   K=j (j<8), j+8 (j>=8)
//                                lane>=16 -> N=lane-16, K=j+8 (j<8), j+16 (j>=8)
__global__ void pack_w2_kernel(const float* __restrict__ w2,
                               unsigned short* __restrict__ wpack) {
  int idx = blockIdx.x * 256 + threadIdx.x;            // 5,308,416 total
  if (idx >= 648 * 16 * 32 * 16) return;
  int j  = idx & 15;
  int l  = (idx >> 4) & 31;
  int nt = (idx >> 9) & 15;
  int kt = idx >> 13;
  int kk = ((j < 8) ? j : j + 8) + ((l < 16) ? 0 : 8);
  int K  = kt * 32 + kk;                                // K = ic*81 + ky*9 + kx
  int oc = nt * 16 + (l & 15);
  wpack[idx] = f2bf(w2[oc * 20736 + K]);
}

// ---------------------------------------------------------------- im2col offsets
// offT[K] = ic*400 + ky*20 + kx
__global__ void offt_kernel(int* __restrict__ offT) {
  int K = blockIdx.x * 256 + threadIdx.x;              // 20736 total
  if (K >= 20736) return;
  int ic = K / 81;
  int r  = K - ic * 81;
  int ky = r / 9;
  int kx = r - ky * 9;
  offT[K] = ic * 400 + ky * 20 + kx;
}

// ---------------------------------------------------------------- conv2 (WMMA)
// Implicit GEMM: M=9216 (b*36+p), N=256 (oc), K=20736, stride-2 conv.
// Block = 256 threads (8 waves). Block tile M=32 x N=256, big K-step = 64.
// wave -> (rg = wave>>2 row-group of 16 rows, n0 = (wave&3)*64): 4 accumulators,
// 8 WMMAs per iteration (2 K-substeps x 4 N-fragments).
// Double-buffered pipeline: iteration t issues TDM(t+1) (32KB weight slab) and
// im2col-gathers A(t+1) into `nxt` LDS buffers while computing from `cur`;
// one s_wait_tensorcnt + one barrier per iteration.
__global__ void conv2_wmma_kernel(const unsigned short* __restrict__ x1,
                                  const unsigned short* __restrict__ wpack,
                                  const int* __restrict__ offT,
                                  const float* __restrict__ b2,
                                  float* __restrict__ out2) {
  const int tid  = threadIdx.x;
  const int wave = tid >> 5;
  const int lane = tid & 31;
  const int m0   = blockIdx.x * 32;                    // 288 blocks
  const int rg   = wave >> 2;                          // row-group 0/1
  const int n0   = (wave & 3) * 64;
  const int nt0  = n0 >> 4;

  __shared__ __align__(128) unsigned short ldsA[2][32 * 64];        // 2 x 4 KB
#if HAVE_TDM
  __shared__ __align__(128) unsigned short ldsB[2][2 * 16 * 32 * 16]; // 2 x 32 KB
#endif

  // gather geometry: thread owns 8 elements of one A row (cols col0..col0+7)
  const int e0   = tid * 8;
  const int grow = e0 >> 6;                            // 0..31
  const int col0 = e0 & 63;
  int rowbase;
  {
    const int mg = m0 + grow;
    const int b  = mg / 36;
    const int p  = mg - b * 36;
    const int oy = p / 6;
    const int ox = p - oy * 6;
    rowbase = b * 102400 + (oy * 2) * 20 + (ox * 2);
  }

  // A-fragment geometry (ISA 7.12.2 16-bit A 16x32):
  // lane<16: chunks at K-local 0 and 16 ; lane>=16: chunks at 8 and 24
  const int arow = rg * 16 + (lane & 15);
  const int acol = (lane >> 4) * 8;

#if HAVE_TDM
  const unsigned int ldsb_off   = (unsigned int)(size_t)(&ldsB[0][0]);
  const unsigned long long wadr = (unsigned long long)(size_t)wpack;
  const i32x4 gz4 = { 0, 0, 0, 0 };
  const i32x8 gz8 = { 0, 0, 0, 0, 0, 0, 0, 0 };
#endif
  const v16bf* wp = reinterpret_cast<const v16bf*>(wpack);

  v8f acc[4] = {{}, {}, {}, {}};

  // ---------------- prologue: stage buffers for ktt = 0 ----------------
#if HAVE_TDM
  if (wave == 0) {
    unsigned long long gaddr = wadr;
    unsigned int g0w3 = (unsigned int)((gaddr >> 32) & 0x1FFFFFFull) | (2u << 30);
    u32x4 g0 = { 1u, ldsb_off, (unsigned int)gaddr, g0w3 };
    i32x8 g1 = { 0x00010000, (int)(16384u << 16), (int)(1u << 16),
                 (int)(16384u << 16), 1, 16384, 0, 0 };
    __builtin_amdgcn_tensor_load_to_lds(g0, g1, gz4, gz4, gz8, 0);
  }
#endif
  {
    const int* op = offT + col0;
    int4 oa = *reinterpret_cast<const int4*>(op);
    int4 ob = *reinterpret_cast<const int4*>(op + 4);
    uint4 pk;
    pk.x = (unsigned)x1[rowbase + oa.x] | ((unsigned)x1[rowbase + oa.y] << 16);
    pk.y = (unsigned)x1[rowbase + oa.z] | ((unsigned)x1[rowbase + oa.w] << 16);
    pk.z = (unsigned)x1[rowbase + ob.x] | ((unsigned)x1[rowbase + ob.y] << 16);
    pk.w = (unsigned)x1[rowbase + ob.z] | ((unsigned)x1[rowbase + ob.w] << 16);
    *reinterpret_cast<uint4*>(&ldsA[0][e0]) = pk;
  }
#if HAVE_TDM
  if (wave == 0) __builtin_amdgcn_s_wait_tensorcnt(0);
#endif
  __syncthreads();

  // ---------------- main pipeline: 324 big K-steps ----------------
  for (int ktt = 0; ktt < 324; ++ktt) {
    const int cur = ktt & 1;
    const int nxt = cur ^ 1;

    if (ktt + 1 < 324) {
#if HAVE_TDM
      if (wave == 0) {
        unsigned long long gaddr = wadr + (unsigned long long)(ktt + 1) * 32768ull;
        unsigned int g0w3 = (unsigned int)((gaddr >> 32) & 0x1FFFFFFull) | (2u << 30);
        u32x4 g0 = { 1u, ldsb_off + (unsigned)nxt * 32768u, (unsigned int)gaddr, g0w3 };
        i32x8 g1 = { 0x00010000, (int)(16384u << 16), (int)(1u << 16),
                     (int)(16384u << 16), 1, 16384, 0, 0 };
        __builtin_amdgcn_tensor_load_to_lds(g0, g1, gz4, gz4, gz8, 0);
      }
#endif
      const int* op = offT + (ktt + 1) * 64 + col0;
      int4 oa = *reinterpret_cast<const int4*>(op);
      int4 ob = *reinterpret_cast<const int4*>(op + 4);
      uint4 pk;
      pk.x = (unsigned)x1[rowbase + oa.x] | ((unsigned)x1[rowbase + oa.y] << 16);
      pk.y = (unsigned)x1[rowbase + oa.z] | ((unsigned)x1[rowbase + oa.w] << 16);
      pk.z = (unsigned)x1[rowbase + ob.x] | ((unsigned)x1[rowbase + ob.y] << 16);
      pk.w = (unsigned)x1[rowbase + ob.z] | ((unsigned)x1[rowbase + ob.w] << 16);
      *reinterpret_cast<uint4*>(&ldsA[nxt][e0]) = pk;
    }

    // ---- compute from cur: 2 K-substeps x 4 N-fragments ----
#pragma unroll
    for (int s = 0; s < 2; ++s) {
      v8bf alo = *reinterpret_cast<const v8bf*>(&ldsA[cur][arow * 64 + s * 32 + acol]);
      v8bf ahi = *reinterpret_cast<const v8bf*>(&ldsA[cur][arow * 64 + s * 32 + acol + 16]);
      v16bf afrag;
#pragma unroll
      for (int i = 0; i < 8; ++i) { afrag[i] = alo[i]; afrag[i + 8] = ahi[i]; }

#pragma unroll
      for (int i = 0; i < 4; ++i) {
#if HAVE_TDM
        v16bf bf = *reinterpret_cast<const v16bf*>(
            &ldsB[cur][((s * 16 + nt0 + i) * 32 + lane) * 16]);
#else
        v16bf bf = wp[((ktt * 2 + s) * 16 + nt0 + i) * 32 + lane];
#endif
        acc[i] = __builtin_amdgcn_wmma_f32_16x16x32_bf16(false, afrag, false, bf,
                                                         (short)0, acc[i], false, false);
      }
    }

#if HAVE_TDM
    if (wave == 0) __builtin_amdgcn_s_wait_tensorcnt(0);
#endif
    __syncthreads();
  }

  // Epilogue: C layout (16x16 f32): vgpr r, lane<16 -> (M=r,N=lane);
  // lane>=16 -> (M=r+8,N=lane-16). Add bias, scatter to (B,256,36).
  const int nn = lane & 15;
  const int mh = (lane >> 4) * 8;
#pragma unroll
  for (int r = 0; r < 8; ++r) {
    int mg = m0 + rg * 16 + mh + r;
    int b  = mg / 36;
    int p  = mg - b * 36;
#pragma unroll
    for (int i = 0; i < 4; ++i) {
      int oc = n0 + i * 16 + nn;
      out2[(b * 256 + oc) * 36 + p] = acc[i][r] + b2[oc];
    }
  }
}

// ---------------------------------------------------------------- routing
// One block per (b,o). u_hat[n,k] computed on the fly; thread (k=tid&15,
// c=tid>>4) holds u for n=c+16t, t=0..71 in registers. Routing collapses to
// b_ij = alpha * u_hat with scalar alpha per (b,o): a1=0, a2=sv1, a3=sv1+sv2.
__global__ void routing_kernel(const float* __restrict__ out2,
                               const float* __restrict__ W,
                               float* __restrict__ vout) {
  const int bo  = blockIdx.x;               // 2560
  const int b   = bo / 10;
  const int o   = bo - b * 10;
  const int tid = threadIdx.x;
  const int k   = tid & 15;
  const int c   = tid >> 4;

  __shared__ float xloc[1152 * 8];          // 36.9 KB
  __shared__ float red[256];
  __shared__ float sbuf[16];
  __shared__ float mbuf[16];
  __shared__ float zbuf[16];
  __shared__ float coef_sh;

  for (int e = tid; e < 9216; e += 256) {
    int n = e >> 3, d = e & 7;
    xloc[e] = out2[((b << 8) + (d << 5) + n / 36) * 36 + (n % 36)];
  }
  __syncthreads();

  float ureg[72];
#pragma unroll
  for (int t = 0; t < 72; ++t) {
    int n = c + (t << 4);
    const float* wrow = W + ((n * 10 + o) * 16 + k) * 8;
    const float* xrow = xloc + n * 8;
    float a = 0.0f;
#pragma unroll
    for (int d = 0; d < 8; ++d) a = fmaf(wrow[d], xrow[d], a);
    ureg[t] = a;
  }

  // ---- iteration 1: uniform c_ij = 1/1152 ----
  {
    float part = 0.0f;
#pragma unroll
    for (int t = 0; t < 72; ++t) part += ureg[t];
    red[tid] = part;
    __syncthreads();
    if (tid < 16) {
      float s = 0.0f;
      for (int cc = 0; cc < 16; ++cc) s += red[cc * 16 + tid];
      sbuf[tid] = s * (1.0f / 1152.0f);
    }
    __syncthreads();
    if (tid == 0) {
      float sn = 0.0f, ss = 0.0f;
      for (int kk = 0; kk < 16; ++kk) { sn += sbuf[kk] * sbuf[kk]; ss += sbuf[kk]; }
      float coef = sn / ((1.0f + sn) * sqrtf(sn));
      coef_sh = coef * ss;                  // alpha2 = sv1
    }
    __syncthreads();
  }
  float alpha = coef_sh;

  // ---- iterations 2 and 3: softmax_n(alpha*u), weighted sum, squash ----
#pragma unroll 1
  for (int iter = 0; iter < 2; ++iter) {
    float pm = -3.4e38f;
#pragma unroll
    for (int t = 0; t < 72; ++t) pm = fmaxf(pm, alpha * ureg[t]);
    red[tid] = pm;
    __syncthreads();
    if (tid < 16) {
      float m = -3.4e38f;
      for (int cc = 0; cc < 16; ++cc) m = fmaxf(m, red[cc * 16 + tid]);
      mbuf[tid] = m;
    }
    __syncthreads();
    float mk = mbuf[k];

    float pz = 0.0f, ps = 0.0f;
#pragma unroll
    for (int t = 0; t < 72; ++t) {
      float e = __expf(alpha * ureg[t] - mk);
      pz += e;
      ps = fmaf(ureg[t], e, ps);
    }
    red[tid] = pz;
    __syncthreads();
    if (tid < 16) {
      float z = 0.0f;
      for (int cc = 0; cc < 16; ++cc) z += red[cc * 16 + tid];
      zbuf[tid] = z;
    }
    __syncthreads();
    red[tid] = ps;
    __syncthreads();
    if (tid < 16) {
      float s = 0.0f;
      for (int cc = 0; cc < 16; ++cc) s += red[cc * 16 + tid];
      sbuf[tid] = s / zbuf[tid];            // s_j[k]
    }
    __syncthreads();
    if (tid == 0) {
      float sn = 0.0f, ss = 0.0f;
      for (int kk = 0; kk < 16; ++kk) { sn += sbuf[kk] * sbuf[kk]; ss += sbuf[kk]; }
      float coef = sn / ((1.0f + sn) * sqrtf(sn));
      coef_sh = (iter == 0) ? (alpha + coef * ss)  // alpha3 = sv1 + sv2
                            : coef;                // final squash coefficient
    }
    __syncthreads();
    alpha = coef_sh;
  }

  if (tid < 16) vout[(b * 10 + o) * 16 + tid] = alpha * sbuf[tid];
}

// ---------------------------------------------------------------- launch
extern "C" void kernel_launch(void* const* d_in, const int* in_sizes, int n_in,
                              void* d_out, int out_size, void* d_ws, size_t ws_size,
                              hipStream_t stream) {
  const float* data = (const float*)d_in[0];
  const float* w1   = (const float*)d_in[1];
  const float* b1   = (const float*)d_in[2];
  const float* w2   = (const float*)d_in[3];
  const float* b2   = (const float*)d_in[4];
  const float* W    = (const float*)d_in[5];

  unsigned char* ws = (unsigned char*)d_ws;
  unsigned short* x1    = (unsigned short*)(ws);                       // 52,428,800 B
  unsigned short* wpack = (unsigned short*)(ws + 52428800u);           // 10,616,832 B
  float*          out2  = (float*)(ws + 63045632u);                    //  9,437,184 B
  int*            offT  = (int*)(ws + 72482816u);                      //     82,944 B

  conv1_kernel  <<<102400, 256, 0, stream>>>(data, w1, b1, x1);
  pack_w2_kernel<<<20736,  256, 0, stream>>>(w2, wpack);
  offt_kernel   <<<81,     256, 0, stream>>>(offT);
  conv2_wmma_kernel<<<288, 256, 0, stream>>>(x1, wpack, offT, b2, out2);
  routing_kernel<<<2560,   256, 0, stream>>>(out2, W, (float*)d_out);
}
